// CrossAttentionLayer_13511967113537
// MI455X (gfx1250) — matrix operand order repaired and checked by reading, not compiled, not run on hardware
//
#include <hip/hip_runtime.h>

// ---------------------------------------------------------------------------
// Problem constants (from reference): B=4, LQ=LK=1024, E=1024, H=16, DH=64, F=4096
// ---------------------------------------------------------------------------
#define BB     4
#define LQ     1024
#define LK     1024
#define EE     1024
#define HH     16
#define DH     64
#define FF     4096
#define MROWS  (BB * LQ)          // 4096 rows of activations
#define SCALE  0.125f             // 1/sqrt(64)
#define LDA    72                 // LDS row stride (elements): 64 + 8 pad, 16B aligned

typedef __attribute__((ext_vector_type(16))) __bf16 bf16x16;
typedef __attribute__((ext_vector_type(8)))  float  v8f;

__device__ __forceinline__ v8f wmma_bf16(bf16x16 a, bf16x16 b, v8f c) {
  return __builtin_amdgcn_wmma_f32_16x16x32_bf16(
      /*neg_a=*/false, a, /*neg_b=*/false, b,
      /*c_mod=*/(short)0, c, /*reuse_a=*/false, /*reuse_b=*/false);
}

// A fragment (16x32 bf16, row-major source, ld = row stride in elements).
// Lane l: row m0 + (l&15); VGPR0-3 = K[8*(l>>4) .. +8), VGPR4-7 = +16.  (ISA 7.12.2)
__device__ __forceinline__ bf16x16 load_a_frag(const __bf16* A, int ld, int m0,
                                               int k0, int lane) {
  const __bf16* p = A + (size_t)(m0 + (lane & 15)) * ld + k0 + ((lane >> 4) << 3);
  union { bf16x16 v; uint4 q[2]; } u;
  u.q[0] = *(const uint4*)(p);
  u.q[1] = *(const uint4*)(p + 16);
  return u.v;
}

// B fragment (32x16 bf16) read from Bt stored [N,K] row-major.
// Lane l: col n0 + (l&15); VGPR0-3 = K[k0 + 8*(l>>4) .. +8), VGPR4-7 = +16.
__device__ __forceinline__ bf16x16 load_b_frag(const __bf16* Bt, int ld, int n0,
                                               int k0, int lane) {
  const __bf16* p = Bt + (size_t)(n0 + (lane & 15)) * ld + k0 + ((lane >> 4) << 3);
  union { bf16x16 v; uint4 q[2]; } u;
  u.q[0] = *(const uint4*)(p);
  u.q[1] = *(const uint4*)(p + 16);
  return u.v;
}

// ---------------------------------------------------------------------------
// CDNA5 async global -> LDS copy (ASYNCcnt-tracked, §4 of async/tensor guide).
// VDST VGPR carries per-lane LDS byte address; GV addressing (64-bit vaddr).
// ---------------------------------------------------------------------------
__device__ __forceinline__ void async_cp_b128(unsigned lds_byte, const void* g) {
  unsigned long long ga = (unsigned long long)(uintptr_t)g;
  asm volatile("global_load_async_to_lds_b128 %0, %1, off"
               :: "v"(lds_byte), "v"(ga) : "memory");
}
__device__ __forceinline__ void wait_async0() {
  asm volatile("s_wait_asynccnt 0x0" ::: "memory");
}

// Issue one K-stage (64 wide) of A[64 rows] and B[128 rows] into LDS.
__device__ __forceinline__ void issue_stage(
    const __bf16* __restrict__ A, const __bf16* __restrict__ Bt, int K,
    int bm0, int bn0, int k0, unsigned sAb, unsigned sBb, int tid) {
  #pragma unroll
  for (int i = 0; i < 2; ++i) {          // A: 64x64 bf16 = 512 16B chunks
    int c = tid + 256 * i;
    int r = c >> 3, cc = (c & 7) << 3;
    async_cp_b128(sAb + (unsigned)(r * LDA + cc) * 2,
                  A + (size_t)(bm0 + r) * K + k0 + cc);
  }
  #pragma unroll
  for (int i = 0; i < 4; ++i) {          // B: 128x64 bf16 = 1024 16B chunks
    int c = tid + 256 * i;
    int r = c >> 3, cc = (c & 7) << 3;
    async_cp_b128(sBb + (unsigned)(r * LDA + cc) * 2,
                  Bt + (size_t)(bn0 + r) * K + k0 + cc);
  }
}

// ---------------------------------------------------------------------------
// Weight transpose + fp32 -> bf16:  W[K,N] -> Wt[N,K]
// ---------------------------------------------------------------------------
__global__ __launch_bounds__(256) void transpose_w_kernel(
    const float* __restrict__ W, __bf16* __restrict__ Wt, int K, int N) {
  __shared__ float tile[32][33];
  int n0 = blockIdx.x * 32, k0 = blockIdx.y * 32;
  int tx = threadIdx.x & 31, ty = threadIdx.x >> 5;
  #pragma unroll
  for (int i = ty; i < 32; i += 8)
    tile[i][tx] = W[(size_t)(k0 + i) * N + n0 + tx];
  __syncthreads();
  #pragma unroll
  for (int i = ty; i < 32; i += 8)
    Wt[(size_t)(n0 + i) * K + k0 + tx] = (__bf16)tile[tx][i];
}

// ---------------------------------------------------------------------------
// Elementwise fp32 -> bf16
// ---------------------------------------------------------------------------
__global__ __launch_bounds__(256) void f32_to_bf16_kernel(
    const float* __restrict__ in, __bf16* __restrict__ out, int n) {
  int i = blockIdx.x * 256 + threadIdx.x;
  if (i < n) out[i] = (__bf16)in[i];
}

// ---------------------------------------------------------------------------
// LayerNorm (row = 1024) -> bf16
// ---------------------------------------------------------------------------
__global__ __launch_bounds__(256) void ln_bf16_kernel(
    const float* __restrict__ x, const float* __restrict__ g,
    const float* __restrict__ bta, __bf16* __restrict__ out) {
  int row = blockIdx.x;
  const float* xr = x + (size_t)row * EE;
  int tid = threadIdx.x;
  float lv[4], s = 0.f, ss = 0.f;
  #pragma unroll
  for (int i = 0; i < 4; ++i) {
    float t = xr[tid + 256 * i];
    lv[i] = t; s += t; ss += t * t;
  }
  #pragma unroll
  for (int o = 16; o; o >>= 1) {
    s  += __shfl_xor(s,  o, 32);
    ss += __shfl_xor(ss, o, 32);
  }
  __shared__ float rs[8], rss[8];
  int wave = tid >> 5, lane = tid & 31;
  if (lane == 0) { rs[wave] = s; rss[wave] = ss; }
  __syncthreads();
  if (tid == 0) {
    float a = 0.f, b2 = 0.f;
    #pragma unroll
    for (int i = 0; i < 8; ++i) { a += rs[i]; b2 += rss[i]; }
    rs[0] = a; rss[0] = b2;
  }
  __syncthreads();
  float mean = rs[0]  * (1.f / EE);
  float var  = rss[0] * (1.f / EE) - mean * mean;
  float rinv = rsqrtf(var + 1e-5f);
  #pragma unroll
  for (int i = 0; i < 4; ++i) {
    int c = tid + 256 * i;
    out[(size_t)row * EE + c] = (__bf16)((lv[i] - mean) * rinv * g[c] + bta[c]);
  }
}

// ---------------------------------------------------------------------------
// WMMA GEMM with double-buffered async global->LDS staging.
// Block: 256 thr = 8 waves; tile 64(M) x 128(N); wave = 32x32 (4 accumulators).
// K consumed in stages of 64 (2 x 16x16x32 WMMA per accumulator per stage).
// ---------------------------------------------------------------------------
template <int RELU, int RES, int OUT_BF16>
__global__ __launch_bounds__(256) void gemm_kernel(
    const __bf16* __restrict__ A, const __bf16* __restrict__ Bt,
    const float* __restrict__ bias, const float* __restrict__ res,
    void* __restrict__ out, int M, int N, int K) {
  __shared__ __bf16 sA[2][64 * LDA];
  __shared__ __bf16 sB[2][128 * LDA];
  int tid = threadIdx.x, wave = tid >> 5, lane = tid & 31;
  int bn0 = blockIdx.x * 128, bm0 = blockIdx.y * 64;
  int wm = (wave >> 2) * 32;       // 0 / 32 within tile
  int wn = (wave & 3) * 32;        // 0 / 32 / 64 / 96 within tile
  unsigned sA_base = (unsigned)(uintptr_t)(void*)&sA[0][0];
  unsigned sB_base = (unsigned)(uintptr_t)(void*)&sB[0][0];

  v8f acc00 = {}, acc01 = {}, acc10 = {}, acc11 = {};
  int NT = K >> 6;
  issue_stage(A, Bt, K, bm0, bn0, 0, sA_base, sB_base, tid);

  for (int kt = 0; kt < NT; ++kt) {
    wait_async0();                 // this wave's stage copies have landed
    __syncthreads();               // all waves' copies + previous-stage reads done
    if (kt + 1 < NT)
      issue_stage(A, Bt, K, bm0, bn0, (kt + 1) << 6,
                  sA_base + (unsigned)(((kt + 1) & 1) * (64 * LDA * 2)),
                  sB_base + (unsigned)(((kt + 1) & 1) * (128 * LDA * 2)), tid);
    const __bf16* a_s = &sA[kt & 1][0];
    const __bf16* b_s = &sB[kt & 1][0];
    #pragma unroll
    for (int kk = 0; kk < 64; kk += 32) {
      bf16x16 b0 = load_b_frag(b_s, LDA, wn,      kk, lane);
      bf16x16 b1 = load_b_frag(b_s, LDA, wn + 16, kk, lane);
      bf16x16 a0 = load_a_frag(a_s, LDA, wm,      kk, lane);
      bf16x16 a1 = load_a_frag(a_s, LDA, wm + 16, kk, lane);
      acc00 = wmma_bf16(a0, b0, acc00);
      acc01 = wmma_bf16(a0, b1, acc01);
      acc10 = wmma_bf16(a1, b0, acc10);
      acc11 = wmma_bf16(a1, b1, acc11);
    }
    __syncthreads();               // reads done before this buffer's next refill
  }

  int lane15 = lane & 15, laneh = (lane >> 4) << 3;
  #pragma unroll
  for (int mi = 0; mi < 2; ++mi) {
    #pragma unroll
    for (int ni = 0; ni < 2; ++ni) {
      v8f acc = (mi == 0) ? (ni == 0 ? acc00 : acc01)
                          : (ni == 0 ? acc10 : acc11);
      int n  = bn0 + wn + ni * 16 + lane15;
      int mb = bm0 + wm + mi * 16 + laneh;
      float bn_ = bias[n];
      #pragma unroll
      for (int e = 0; e < 8; ++e) {
        int r = mb + e;
        float v = acc[e] + bn_;
        if (RELU) v = fmaxf(v, 0.f);
        if (RES)  v += res[(size_t)r * N + n];
        if (OUT_BF16) ((__bf16*)out)[(size_t)r * N + n] = (__bf16)v;
        else          ((float*)out)[(size_t)r * N + n] = v;
      }
    }
  }
}

// ---------------------------------------------------------------------------
// V transpose per batch: V[b][k][e] (bf16) -> Vt[b][e][k]   (1024x1024 per b)
// ---------------------------------------------------------------------------
__global__ __launch_bounds__(256) void transpose_v_kernel(
    const __bf16* __restrict__ V, __bf16* __restrict__ Vt) {
  __shared__ __bf16 tile[32][33];
  int b = blockIdx.z;
  int c0 = blockIdx.x * 32, r0 = blockIdx.y * 32;
  int tx = threadIdx.x & 31, ty = threadIdx.x >> 5;
  #pragma unroll
  for (int i = ty; i < 32; i += 8)
    tile[i][tx] = V[((size_t)b * LK + r0 + i) * EE + c0 + tx];
  __syncthreads();
  #pragma unroll
  for (int i = ty; i < 32; i += 8)
    Vt[((size_t)b * EE + c0 + i) * LK + r0 + tx] = tile[tx][i];
}

// ---------------------------------------------------------------------------
// Attention: block = (16-query tile, batch); loop over 16 heads.
//   S = Q Kᵀ (WMMA, all 8 waves) -> LDS f32 [16x1024]
//   shuffle softmax; attn-mean accumulated to global; P -> bf16 in-place in LDS
//   O = P V  (WMMA, waves 0-3, A frag from LDS) -> ctx bf16
// ---------------------------------------------------------------------------
__global__ __launch_bounds__(256) void attn_kernel(
    const __bf16* __restrict__ Q,   // [B*LQ, E]
    const __bf16* __restrict__ Kb,  // [B*LK, E]
    const __bf16* __restrict__ Vt,  // [B*H, DH, LK]
    __bf16* __restrict__ ctx,       // [B*LQ, E]
    float* __restrict__ attn) {     // [B, LQ, LK]
  __shared__ float sS[16 * LK];     // 64 KB; reused as bf16 P
  __bf16* sP = (__bf16*)sS;
  int q0 = blockIdx.x * 16;
  int b  = blockIdx.y;
  int tid = threadIdx.x, wave = tid >> 5, lane = tid & 31;
  int r = tid >> 4, j = tid & 15;   // softmax: 16 threads per score row

  for (int h = 0; h < HH; ++h) {
    const __bf16* Qh = Q + ((size_t)(b * LQ + q0)) * EE + h * DH;
    bf16x16 qa0 = load_a_frag(Qh, EE, 0, 0,  lane);
    bf16x16 qa1 = load_a_frag(Qh, EE, 0, 32, lane);

    // ---- scores: each wave owns 8 key tiles of 16 ----
    for (int kt = wave; kt < LK / 16; kt += 8) {
      const __bf16* Kh = Kb + ((size_t)(b * LK + kt * 16)) * EE + h * DH;
      v8f acc = {};
      bf16x16 kb0 = load_b_frag(Kh, EE, 0, 0,  lane);
      acc = wmma_bf16(qa0, kb0, acc);
      bf16x16 kb1 = load_b_frag(Kh, EE, 0, 32, lane);
      acc = wmma_bf16(qa1, kb1, acc);
      int n = lane & 15, mb = (lane >> 4) << 3;
      #pragma unroll
      for (int e = 0; e < 8; ++e)
        sS[(mb + e) * LK + kt * 16 + n] = acc[e] * SCALE;
    }
    __syncthreads();

    // ---- softmax over 1024 keys (row r, 64 elems per thread) ----
    float v[64], m = -1e30f;
    #pragma unroll
    for (int i = 0; i < 64; ++i) {
      v[i] = sS[r * LK + j + 16 * i];
      m = fmaxf(m, v[i]);
    }
    #pragma unroll
    for (int o = 8; o; o >>= 1) m = fmaxf(m, __shfl_xor(m, o, 32));
    float s = 0.f;
    #pragma unroll
    for (int i = 0; i < 64; ++i) { v[i] = __expf(v[i] - m); s += v[i]; }
    #pragma unroll
    for (int o = 8; o; o >>= 1) s += __shfl_xor(s, o, 32);
    float inv = 1.f / s;

    // ---- head-averaged attention weights straight to d_out ----
    float* ap = attn + ((size_t)(b * LQ + q0 + r)) * LK + j;
    if (h == 0) {
      #pragma unroll
      for (int i = 0; i < 64; ++i) ap[16 * i] = v[i] * inv * (1.f / HH);
    } else {
      #pragma unroll
      for (int i = 0; i < 64; ++i) ap[16 * i] += v[i] * inv * (1.f / HH);
    }
    __syncthreads();                 // all f32 reads of sS done
    #pragma unroll
    for (int i = 0; i < 64; ++i)
      sP[r * LK + j + 16 * i] = (__bf16)(v[i] * inv);
    __syncthreads();

    // ---- O = P * V : waves 0..3 each own 16 of the 64 head dims ----
    if (wave < 4) {
      const __bf16* Vh = Vt + (size_t)(b * HH + h) * DH * LK;  // [DH, LK]
      v8f acc = {};
      #pragma unroll 4
      for (int k0 = 0; k0 < LK; k0 += 32) {
        bf16x16 pa = load_a_frag(sP, LK, 0, k0, lane);         // ds_load_b128
        bf16x16 vb = load_b_frag(Vh, LK, wave * 16, k0, lane);
        acc = wmma_bf16(pa, vb, acc);
      }
      int n = lane & 15, mb = (lane >> 4) << 3;
      __bf16* cp = ctx + ((size_t)(b * LQ + q0)) * EE + h * DH + wave * 16 + n;
      #pragma unroll
      for (int e = 0; e < 8; ++e)
        cp[(size_t)(mb + e) * EE] = (__bf16)acc[e];
    }
    __syncthreads();
  }
}

// ---------------------------------------------------------------------------
// Host launcher
// ---------------------------------------------------------------------------
extern "C" void kernel_launch(void* const* d_in, const int* in_sizes, int n_in,
                              void* d_out, int out_size, void* d_ws, size_t ws_size,
                              hipStream_t stream) {
  (void)in_sizes; (void)n_in; (void)out_size; (void)ws_size;
  const float* tgt    = (const float*)d_in[0];
  const float* memory = (const float*)d_in[1];
  const float* ln1_g  = (const float*)d_in[2];
  const float* ln1_b  = (const float*)d_in[3];
  const float* wq = (const float*)d_in[4];  const float* bq = (const float*)d_in[5];
  const float* wk = (const float*)d_in[6];  const float* bk = (const float*)d_in[7];
  const float* wv = (const float*)d_in[8];  const float* bv = (const float*)d_in[9];
  const float* wo = (const float*)d_in[10]; const float* bo = (const float*)d_in[11];
  const float* ln3_g  = (const float*)d_in[12];
  const float* ln3_b  = (const float*)d_in[13];
  const float* w1 = (const float*)d_in[14]; const float* b1 = (const float*)d_in[15];
  const float* w2 = (const float*)d_in[16]; const float* b2 = (const float*)d_in[17];

  float* out_x    = (float*)d_out;                       // [B,LQ,E]
  float* out_attn = (float*)d_out + (size_t)MROWS * EE;  // [B,LQ,LK]

  char* ws = (char*)d_ws;
  auto alloc = [&](size_t bytes) -> void* {
    void* p = (void*)ws;
    ws += (bytes + 255) & ~(size_t)255;
    return p;
  };
  __bf16* wqT = (__bf16*)alloc((size_t)EE * EE * 2);
  __bf16* wkT = (__bf16*)alloc((size_t)EE * EE * 2);
  __bf16* wvT = (__bf16*)alloc((size_t)EE * EE * 2);
  __bf16* woT = (__bf16*)alloc((size_t)EE * EE * 2);
  __bf16* w1T = (__bf16*)alloc((size_t)EE * FF * 2);
  __bf16* w2T = (__bf16*)alloc((size_t)FF * EE * 2);
  __bf16* t2   = (__bf16*)alloc((size_t)MROWS * EE * 2);
  __bf16* memb = (__bf16*)alloc((size_t)MROWS * EE * 2);
  __bf16* qb   = (__bf16*)alloc((size_t)MROWS * EE * 2);
  __bf16* kb   = (__bf16*)alloc((size_t)MROWS * EE * 2);
  __bf16* vb   = (__bf16*)alloc((size_t)MROWS * EE * 2);
  __bf16* vt   = (__bf16*)alloc((size_t)MROWS * EE * 2);
  __bf16* ctx  = (__bf16*)alloc((size_t)MROWS * EE * 2);
  float*  xbuf = (float*)alloc((size_t)MROWS * EE * 4);
  __bf16* t3   = (__bf16*)alloc((size_t)MROWS * EE * 2);
  __bf16* h1   = (__bf16*)alloc((size_t)MROWS * FF * 2);

  dim3 blk(256);

  // Weight transpose + bf16 convert
  transpose_w_kernel<<<dim3(EE / 32, EE / 32), blk, 0, stream>>>(wq, wqT, EE, EE);
  transpose_w_kernel<<<dim3(EE / 32, EE / 32), blk, 0, stream>>>(wk, wkT, EE, EE);
  transpose_w_kernel<<<dim3(EE / 32, EE / 32), blk, 0, stream>>>(wv, wvT, EE, EE);
  transpose_w_kernel<<<dim3(EE / 32, EE / 32), blk, 0, stream>>>(wo, woT, EE, EE);
  transpose_w_kernel<<<dim3(FF / 32, EE / 32), blk, 0, stream>>>(w1, w1T, EE, FF);
  transpose_w_kernel<<<dim3(EE / 32, FF / 32), blk, 0, stream>>>(w2, w2T, FF, EE);

  // Activation conversions
  f32_to_bf16_kernel<<<(MROWS * EE) / 256, blk, 0, stream>>>(memory, memb, MROWS * EE);
  ln_bf16_kernel<<<MROWS, blk, 0, stream>>>(tgt, ln1_g, ln1_b, t2);

  // Q / K / V projections (bf16 out), tile 64x128
  gemm_kernel<0, 0, 1><<<dim3(EE / 128, MROWS / 64), blk, 0, stream>>>(
      t2, wqT, bq, nullptr, qb, MROWS, EE, EE);
  gemm_kernel<0, 0, 1><<<dim3(EE / 128, MROWS / 64), blk, 0, stream>>>(
      memb, wkT, bk, nullptr, kb, MROWS, EE, EE);
  gemm_kernel<0, 0, 1><<<dim3(EE / 128, MROWS / 64), blk, 0, stream>>>(
      memb, wvT, bv, nullptr, vb, MROWS, EE, EE);

  // V -> [B,H,DH,LK]
  transpose_v_kernel<<<dim3(EE / 32, LK / 32, BB), blk, 0, stream>>>(vb, vt);

  // Attention (also writes head-averaged weights to out_attn)
  attn_kernel<<<dim3(LQ / 16, BB), blk, 0, stream>>>(qb, kb, vt, ctx, out_attn);

  // x = tgt + ctx @ wo + bo   (f32)
  gemm_kernel<0, 1, 0><<<dim3(EE / 128, MROWS / 64), blk, 0, stream>>>(
      ctx, woT, bo, tgt, xbuf, MROWS, EE, EE);

  // t3 = LN3(x) -> bf16
  ln_bf16_kernel<<<MROWS, blk, 0, stream>>>(xbuf, ln3_g, ln3_b, t3);

  // h1 = relu(t3 @ w1 + b1) -> bf16
  gemm_kernel<1, 0, 1><<<dim3(FF / 128, MROWS / 64), blk, 0, stream>>>(
      t3, w1T, b1, nullptr, h1, MROWS, FF, EE);

  // out = x + h1 @ w2 + b2   (f32, final output)
  gemm_kernel<0, 1, 0><<<dim3(EE / 128, MROWS / 64), blk, 0, stream>>>(
      h1, w2T, b2, xbuf, out_x, MROWS, EE, FF);
}
